// Block_9268539425531
// MI455X (gfx1250) — matrix operand order repaired and checked by reading, compiled, not run on hardware
//
#include <hip/hip_runtime.h>
#include <hip/hip_bf16.h>

// ---------------------------------------------------------------------------
// MLA transformer block for MI455X (gfx1250, wave32, WMMA).
// B=2, T=2048, C=2048, NH=16, DK=128, DHR=64, LAT=512. R = B*T = 4096 rows.
// All heavy math runs through v_wmma_f32_16x16x32_f16 (f16 in / f32 accum).
// GEMMs: 128x128 block tile, double-buffered LDS, fused bias/GELU/residual,
// optional f16 output (feeds the next GEMM without an fp32 round trip).
// ---------------------------------------------------------------------------

typedef __attribute__((ext_vector_type(16))) _Float16 v16h;
typedef __attribute__((ext_vector_type(8)))  float    v8f;

#define FLAG_GELU  1
#define FLAG_RESID 2

__device__ __forceinline__ v8f zero8() {
  v8f z = {0.f, 0.f, 0.f, 0.f, 0.f, 0.f, 0.f, 0.f};
  return z;
}

__device__ __forceinline__ v8f wmma16(const v16h& a, const v16h& b, const v8f& c) {
  return __builtin_amdgcn_wmma_f32_16x16x32_f16(false, a, false, b, (short)0, c,
                                                false, false);
}

// gelu(tanh approx): 0.5*u*(1+tanh(t)) == u * sigmoid(2t).
// v_rcp_f32 instead of IEEE division keeps the epilogue straight-line VALU.
__device__ __forceinline__ float gelu_fast(float u) {
  float t = 0.7978845608028654f * (u + 0.044715f * u * u * u);
  return u * __builtin_amdgcn_rcpf(1.0f + __expf(-2.0f * t));
}

// ------------------------------- casts -------------------------------------

__global__ void cast_f32_f16(const float* __restrict__ src,
                             _Float16* __restrict__ dst, long n) {
  long i = (long)blockIdx.x * blockDim.x + threadIdx.x;
  long stride = (long)gridDim.x * blockDim.x;
  for (; i < n; i += stride) dst[i] = (_Float16)src[i];
}

// ----------------------------- layernorm -----------------------------------

__global__ __launch_bounds__(256) void layernorm_f16(
    const float* __restrict__ x, const float* __restrict__ g,
    const float* __restrict__ bb, _Float16* __restrict__ out, int Cd) {
  __shared__ float s1[256];
  __shared__ float s2[256];
  const int row = blockIdx.x;
  const int tid = threadIdx.x;
  const float* xr = x + (long)row * Cd;
  float a = 0.f, q = 0.f;
  for (int c = tid; c < Cd; c += 256) {
    float v = xr[c];
    a += v; q += v * v;
  }
  s1[tid] = a; s2[tid] = q;
  __syncthreads();
  for (int s = 128; s > 0; s >>= 1) {
    if (tid < s) { s1[tid] += s1[tid + s]; s2[tid] += s2[tid + s]; }
    __syncthreads();
  }
  float mean = s1[0] / (float)Cd;
  float var  = s2[0] / (float)Cd - mean * mean;
  float rstd = rsqrtf(var + 1e-6f);
  _Float16* orow = out + (long)row * Cd;
  for (int c = tid; c < Cd; c += 256)
    orow[c] = (_Float16)((xr[c] - mean) * rstd * g[c] + bb[c]);
}

// -------------------------------- RoPE -------------------------------------
// x[row, c] with row = b*2048 + t.  freq = t+1; theta = 10000^(-2*(c/2)/d).
// out[2i] = x0*cos - x1*sin ; out[2i+1] = x1*cos + x0*sin  (in-place).

__global__ void rope_inplace(float* __restrict__ x, int d, long rows) {
  const int half = d >> 1;
  long npairs = rows * half;
  long i = (long)blockIdx.x * blockDim.x + threadIdx.x;
  long stride = (long)gridDim.x * blockDim.x;
  for (; i < npairs; i += stride) {
    long row = i / half;
    int  pi  = (int)(i - row * half);
    int  c0  = pi * 2;
    float t     = (float)(row & 2047) + 1.0f;          // row % T, T=2048
    float theta = __expf(-2.0f * (float)pi / (float)d * 9.210340371976184f);
    float ang = t * theta;
    float cs = cosf(ang), sn = sinf(ang);
    float* px = x + row * d + c0;
    float x0 = px[0], x1 = px[1];
    px[0] = x0 * cs - x1 * sn;
    px[1] = x1 * cs + x0 * sn;
  }
}

// ------------------------- 128x128 WMMA GEMM -------------------------------
// C[M,N] = A[M,K] (f16, row stride lda) @ B[K,N] (f16) + bias,
// optional GELU / residual; writes f32 (Cout) and/or f16 (Cout16).
// 8 waves in a 4x2 grid; each wave owns 32x64 = 2x4 WMMA tiles.
// Double-buffered LDS: one barrier per 32-deep K step, 8 WMMAs per step.

__global__ __launch_bounds__(256) void gemm128_f16_wmma(
    const _Float16* __restrict__ A, int lda, const _Float16* __restrict__ Bw,
    const float* __restrict__ bias, float* __restrict__ Cout,
    _Float16* __restrict__ Cout16, const float* __restrict__ resid,
    int M, int N, int K, int flags) {
  __shared__ _Float16 As[2][128][40];   // 128 x 32 per buffer (pad 40)
  __shared__ _Float16 Bs[2][32][136];   // 32 x 128 per buffer (pad 136)
  const int tid  = threadIdx.x;
  const int lane = tid & 31;
  const int w    = tid >> 5;
  const int wr   = w >> 1;              // 0..3 -> M offset 32*wr
  const int wc   = w & 1;               // 0..1 -> N offset 64*wc
  const int bm = blockIdx.y * 128, bn = blockIdx.x * 128;

  v8f acc[2][4];
#pragma unroll
  for (int mi = 0; mi < 2; ++mi)
#pragma unroll
    for (int j = 0; j < 4; ++j) acc[mi][j] = zero8();

  // Tile staging: A 128x32 halves, B 32x128 halves; 16 halves/thread each.
  const int ar = tid >> 1, ac = (tid & 1) * 16;
  const int br = tid >> 3, bc = (tid & 7) * 16;
  uint4 ra0, ra1, rb0, rb1;

  auto gload = [&](int kb) {
    const _Float16* ap = &A[(long)(bm + ar) * lda + kb + ac];
    const _Float16* bp = &Bw[(long)(kb + br) * N + bn + bc];
    ra0 = *(const uint4*)(ap);
    ra1 = *(const uint4*)(ap + 8);
    rb0 = *(const uint4*)(bp);
    rb1 = *(const uint4*)(bp + 8);
  };
  auto sstore = [&](int buf) {
    *(uint4*)(&As[buf][ar][ac])     = ra0;
    *(uint4*)(&As[buf][ar][ac + 8]) = ra1;
    *(uint4*)(&Bs[buf][br][bc])     = rb0;
    *(uint4*)(&Bs[buf][br][bc + 8]) = rb1;
  };

  gload(0);
  sstore(0);
  __syncthreads();

  int cur = 0;
  for (int kb = 0; kb < K; kb += 32) {
    const bool has_next = (kb + 32) < K;
    if (has_next) {
      gload(kb + 32);
      __builtin_prefetch(&A[(long)(bm + ar) * lda + kb + 64 + ac], 0, 0);
      __builtin_prefetch(&Bw[(long)(kb + 64 + br) * N + bn + bc], 0, 0);
    }

    // A fragments: lane L -> M = base + L%16; half h -> K per CDNA5 layout.
    v16h af[2], bf[4];
#pragma unroll
    for (int mi = 0; mi < 2; ++mi) {
      const int am = 32 * wr + 16 * mi + (lane & 15);
#pragma unroll
      for (int h = 0; h < 16; ++h) {
        int kk = (h < 8 ? h : h + 8) + ((lane & 16) ? 8 : 0);
        af[mi][h] = As[cur][am][kk];
      }
    }
    // B fragments: lane L -> N = base + L%16; half h -> K per CDNA5 layout.
#pragma unroll
    for (int j = 0; j < 4; ++j) {
#pragma unroll
      for (int h = 0; h < 16; ++h) bf[j][h] = Bs[cur][(h < 8 ? h : h + 8) + ((lane & 16) ? 8 : 0)][64 * wc + 16 * j + (lane & 15)];
    }
#pragma unroll
    for (int mi = 0; mi < 2; ++mi)
#pragma unroll
      for (int j = 0; j < 4; ++j) acc[mi][j] = wmma16(af[mi], bf[j], acc[mi][j]);

    if (has_next) sstore(cur ^ 1);
    __syncthreads();
    cur ^= 1;
  }

  // Epilogue: C/D layout lane L -> N = L%16, reg r -> M = r + (L>=16 ? 8 : 0).
  const int mhi = (lane & 16) ? 8 : 0;
  const int nl  = lane & 15;
#pragma unroll
  for (int mi = 0; mi < 2; ++mi) {
#pragma unroll
    for (int j = 0; j < 4; ++j) {
#pragma unroll
      for (int r = 0; r < 8; ++r) {
        int row = bm + 32 * wr + 16 * mi + r + mhi;
        int col = bn + 64 * wc + 16 * j + nl;
        float v = acc[mi][j][r] + bias[col];
        if (flags & FLAG_GELU)  v = gelu_fast(v);
        if (flags & FLAG_RESID) v += resid[(long)row * N + col];
        if (Cout)   Cout[(long)row * N + col]   = v;
        if (Cout16) Cout16[(long)row * N + col] = (_Float16)v;
      }
    }
  }
}

// ------------------------- 64x64 WMMA GEMM (small N) -----------------------
// Used only for the N=64 w_kr projection.

__global__ __launch_bounds__(256) void gemm64_f16_wmma(
    const _Float16* __restrict__ A, int lda, const _Float16* __restrict__ Bw,
    const float* __restrict__ bias, float* __restrict__ Cout,
    int M, int N, int K) {
  __shared__ _Float16 As[64][40];
  __shared__ _Float16 Bs[32][72];
  const int tid  = threadIdx.x;
  const int lane = tid & 31;
  const int w    = tid >> 5;
  const int wr   = w >> 1;
  const int wc   = w & 1;
  const int bm = blockIdx.y * 64, bn = blockIdx.x * 64;

  v8f acc0 = zero8(), acc1 = zero8();
  const int arow = tid >> 2, ac0 = (tid & 3) * 8;
  const int brow = tid >> 3, bc0 = (tid & 7) * 8;

  for (int kb = 0; kb < K; kb += 32) {
    *(uint4*)(&As[arow][ac0]) =
        *(const uint4*)(&A[(long)(bm + arow) * lda + kb + ac0]);
    *(uint4*)(&Bs[brow][bc0]) =
        *(const uint4*)(&Bw[(long)(kb + brow) * N + bn + bc0]);
    if (kb + 32 < K) {
      __builtin_prefetch(&A[(long)(bm + arow) * lda + kb + 32 + ac0], 0, 0);
      __builtin_prefetch(&Bw[(long)(kb + 32 + brow) * N + bn + bc0], 0, 0);
    }
    __syncthreads();

    v16h a, b0, b1;
    const int am = 16 * wr + (lane & 15);
#pragma unroll
    for (int h = 0; h < 16; ++h) {
      int kk = (h < 8 ? h : h + 8) + ((lane & 16) ? 8 : 0);
      a[h] = As[am][kk];
    }
#pragma unroll
    for (int h = 0; h < 16; ++h) {
      int kk = (h < 8 ? h : h + 8) + ((lane & 16) ? 8 : 0);
      b0[h] = Bs[kk][32 * wc + (lane & 15)];
      b1[h] = Bs[kk][32 * wc + 16 + (lane & 15)];
    }
    acc0 = wmma16(a, b0, acc0);
    acc1 = wmma16(a, b1, acc1);
    __syncthreads();
  }

  const int mhi = (lane & 16) ? 8 : 0;
  const int nl  = lane & 15;
#pragma unroll
  for (int r = 0; r < 8; ++r) {
    int row = bm + 16 * wr + r + mhi;
    int c0  = bn + 32 * wc + nl;
    int c1  = c0 + 16;
    Cout[(long)row * N + c0] = acc0[r] + bias[c0];
    Cout[(long)row * N + c1] = acc1[r] + bias[c1];
  }
}

// ------------------------ flash attention (MLA) ----------------------------
// Per (64-row Q tile, head, batch). Q/K are 192-wide (DK=128 ++ DHR=64),
// V is 128-wide. Online softmax, both matmuls via WMMA f16.

__global__ __launch_bounds__(256) void mla_flash_attn(
    const _Float16* __restrict__ qb,   // [R,2048]  q heads: col nh*128+d
    const _Float16* __restrict__ qr,   // [R,1024]  q rope:  col nh*64+d
    const _Float16* __restrict__ kvb,  // [R,4096]  k: 0..2047, v: 2048..4095
    const _Float16* __restrict__ kr,   // [R,64]    shared k rope
    _Float16* __restrict__ out) {      // [R,2048]  col nh*128+d
  __shared__ _Float16 Qs[64][200];
  __shared__ _Float16 Ks[64][200];
  __shared__ _Float16 Vs[64][136];
  __shared__ float    Sf[64][68];
  __shared__ _Float16 Ps[64][72];
  __shared__ float mrow[64], lrow[64], srow[64];

  const int tid = threadIdx.x, lane = tid & 31, w = tid >> 5;
  const int wr = w >> 1, wc = w & 1;
  const int qt = blockIdx.x, nh = blockIdx.y, bz = blockIdx.z;
  const long rbase = (long)bz * 2048;
  const int  q0 = qt * 64;
  const float rscale = 0.08838834764831845f;   // 1/sqrt(DK=128)

  for (int i = tid; i < 64 * 192; i += 256) {
    int r = i / 192, c = i - r * 192;
    long grow = rbase + q0 + r;
    Qs[r][c] = (c < 128) ? qb[grow * 2048 + nh * 128 + c]
                         : qr[grow * 1024 + nh * 64 + (c - 128)];
  }
  if (tid < 64) { mrow[tid] = -3.0e38f; lrow[tid] = 0.0f; }

  v8f o0 = zero8(), o1 = zero8(), o2 = zero8(), o3 = zero8();

  for (int kt = 0; kt <= qt; ++kt) {
    const int k0 = kt * 64;
    for (int i = tid; i < 64 * 192; i += 256) {
      int r = i / 192, c = i - r * 192;
      long grow = rbase + k0 + r;
      Ks[r][c] = (c < 128) ? kvb[grow * 4096 + nh * 128 + c]
                           : kr[grow * 64 + (c - 128)];
    }
    for (int i = tid; i < 64 * 128; i += 256) {
      int r = i >> 7, c = i & 127;
      long grow = rbase + k0 + r;
      Vs[r][c] = kvb[grow * 4096 + 2048 + nh * 128 + c];
    }
    __syncthreads();

    // ---- S = (Q . K^T) * rscale, causal-masked ----
    v8f s0 = zero8(), s1 = zero8();
    const int am = 16 * wr + (lane & 15);
#pragma unroll
    for (int kc = 0; kc < 192; kc += 32) {
      v16h a, b0, b1;
#pragma unroll
      for (int h = 0; h < 16; ++h) {
        int kk = (h < 8 ? h : h + 8) + ((lane & 16) ? 8 : 0);
        a[h] = Qs[am][kc + kk];
      }
#pragma unroll
      for (int h = 0; h < 16; ++h) {          // B = K^T: B[k][n] = K[n][k]
        int kk = (h < 8 ? h : h + 8) + ((lane & 16) ? 8 : 0);
        b0[h] = Ks[32 * wc + (lane & 15)][kc + kk];
        b1[h] = Ks[32 * wc + 16 + (lane & 15)][kc + kk];
      }
      s0 = wmma16(a, b0, s0);
      s1 = wmma16(a, b1, s1);
    }
    {
      const int mhi = (lane & 16) ? 8 : 0;
      const int nl  = lane & 15;
#pragma unroll
      for (int r = 0; r < 8; ++r) {
        int row = 16 * wr + r + mhi;
        int c0 = 32 * wc + nl, c1 = c0 + 16;
        float v0 = s0[r] * rscale, v1 = s1[r] * rscale;
        if (k0 + c0 > q0 + row) v0 = -9.0e15f;
        if (k0 + c1 > q0 + row) v1 = -9.0e15f;
        Sf[row][c0] = v0;
        Sf[row][c1] = v1;
      }
    }
    __syncthreads();

    // ---- online softmax (one thread per row) ----
    if (tid < 64) {
      int row = tid;
      float mold = mrow[row], mx = mold;
#pragma unroll 8
      for (int c = 0; c < 64; ++c) mx = fmaxf(mx, Sf[row][c]);
      float sc = __expf(mold - mx);
      float l  = lrow[row] * sc;
#pragma unroll 8
      for (int c = 0; c < 64; ++c) {
        float pv = __expf(Sf[row][c] - mx);
        l += pv;
        Ps[row][c] = (_Float16)pv;
      }
      mrow[row] = mx; lrow[row] = l; srow[row] = sc;
    }
    __syncthreads();

    // ---- rescale O, then O += P . V ----
    {
      const int mhi = (lane & 16) ? 8 : 0;
#pragma unroll
      for (int r = 0; r < 8; ++r) {
        float sc = srow[16 * wr + r + mhi];
        o0[r] *= sc; o1[r] *= sc; o2[r] *= sc; o3[r] *= sc;
      }
    }
#pragma unroll
    for (int kc = 0; kc < 64; kc += 32) {
      v16h a, b0, b1, b2, b3;
#pragma unroll
      for (int h = 0; h < 16; ++h) {
        int kk = (h < 8 ? h : h + 8) + ((lane & 16) ? 8 : 0);
        a[h] = Ps[am][kc + kk];
      }
#pragma unroll
      for (int h = 0; h < 16; ++h) {
        int kk = (h < 8 ? h : h + 8) + ((lane & 16) ? 8 : 0);
        b0[h] = Vs[kc + kk][64 * wc + (lane & 15)];
        b1[h] = Vs[kc + kk][64 * wc + 16 + (lane & 15)];
        b2[h] = Vs[kc + kk][64 * wc + 32 + (lane & 15)];
        b3[h] = Vs[kc + kk][64 * wc + 48 + (lane & 15)];
      }
      o0 = wmma16(a, b0, o0);
      o1 = wmma16(a, b1, o1);
      o2 = wmma16(a, b2, o2);
      o3 = wmma16(a, b3, o3);
    }
    __syncthreads();
  }

  // ---- epilogue: O / l, write heads back as [R, NH*DK] f16 ----
  const int mhi = (lane & 16) ? 8 : 0;
  const int nl  = lane & 15;
#pragma unroll
  for (int r = 0; r < 8; ++r) {
    int row = 16 * wr + r + mhi;
    float linv = __builtin_amdgcn_rcpf(lrow[row]);
    long grow = rbase + q0 + row;
    _Float16* po = out + grow * 2048 + nh * 128 + 64 * wc + nl;
    po[0]  = (_Float16)(o0[r] * linv);
    po[16] = (_Float16)(o1[r] * linv);
    po[32] = (_Float16)(o2[r] * linv);
    po[48] = (_Float16)(o3[r] * linv);
  }
}

// ------------------------------ launcher -----------------------------------

extern "C" void kernel_launch(void* const* d_in, const int* in_sizes, int n_in,
                              void* d_out, int out_size, void* d_ws, size_t ws_size,
                              hipStream_t stream) {
  (void)in_sizes; (void)n_in; (void)out_size; (void)ws_size;
  const float* x         = (const float*)d_in[0];
  const float* ln1_scale = (const float*)d_in[1];
  const float* ln1_bias  = (const float*)d_in[2];
  const float* w_down    = (const float*)d_in[3];
  const float* b_down    = (const float*)d_in[4];
  const float* w_kr      = (const float*)d_in[5];
  const float* b_kr      = (const float*)d_in[6];
  const float* w_qr      = (const float*)d_in[7];
  const float* b_qr      = (const float*)d_in[8];
  const float* w_ukv     = (const float*)d_in[9];
  const float* b_ukv     = (const float*)d_in[10];
  const float* w_uq      = (const float*)d_in[11];
  const float* b_uq      = (const float*)d_in[12];
  const float* w_o       = (const float*)d_in[13];
  const float* b_o       = (const float*)d_in[14];
  const float* ln2_scale = (const float*)d_in[15];
  const float* ln2_bias  = (const float*)d_in[16];
  const float* w_ff1     = (const float*)d_in[17];
  const float* b_ff1     = (const float*)d_in[18];
  const float* w_ff2     = (const float*)d_in[19];
  const float* b_ff2     = (const float*)d_in[20];
  float* out = (float*)d_out;

  const long R = 4096;  // B*T

  // -------- workspace carve --------
  char* p = (char*)d_ws;
  auto take = [&](size_t bytes) -> void* {
    void* r = (void*)p;
    p += (bytes + 255) & ~(size_t)255;
    return r;
  };
  _Float16* wdown16 = (_Float16*)take(2048L * 1024 * 2);
  _Float16* wkr16   = (_Float16*)take(2048L * 64 * 2);
  _Float16* wqr16   = (_Float16*)take(2048L * 1024 * 2);
  _Float16* wukv16  = (_Float16*)take(512L * 4096 * 2);
  _Float16* wuq16   = (_Float16*)take(512L * 2048 * 2);
  _Float16* wo16    = (_Float16*)take(2048L * 2048 * 2);
  _Float16* wff1x   = (_Float16*)take(2048L * 8192 * 2);
  _Float16* wff2x   = (_Float16*)take(8192L * 2048 * 2);
  _Float16* h16     = (_Float16*)take(R * 2048 * 2);
  _Float16* dkv16   = (_Float16*)take(R * 1024 * 2);   // [ckv | cq] f16
  float*    krlin   = (float*)   take(R * 64 * 4);
  _Float16* kr16    = (_Float16*)take(R * 64 * 2);
  float*    qrlin   = (float*)   take(R * 1024 * 4);
  _Float16* qr16    = (_Float16*)take(R * 1024 * 2);
  _Float16* kv16    = (_Float16*)take(R * 4096 * 2);
  _Float16* qb16    = (_Float16*)take(R * 2048 * 2);
  _Float16* attn16  = (_Float16*)take(R * 2048 * 2);
  float*    x1      = (float*)   take(R * 2048 * 4);
  _Float16* h2_16   = (_Float16*)take(R * 2048 * 2);
  _Float16* ff1_16  = (_Float16*)take(R * 8192 * 2);

  auto castN = [&](const float* s, _Float16* d, long n) {
    cast_f32_f16<<<4096, 256, 0, stream>>>(s, d, n);
  };
  auto gemm = [&](const _Float16* A, int lda, const _Float16* Bw,
                  const float* bias, float* C, _Float16* C16,
                  const float* resid, int M, int N, int K, int flags) {
    dim3 g(N / 128, M / 128);
    gemm128_f16_wmma<<<g, 256, 0, stream>>>(A, lda, Bw, bias, C, C16, resid,
                                            M, N, K, flags);
  };

  // -------- weight casts (f32 -> f16) --------
  castN(w_down, wdown16, 2048L * 1024);
  castN(w_kr,   wkr16,   2048L * 64);
  castN(w_qr,   wqr16,   2048L * 1024);
  castN(w_ukv,  wukv16,  512L * 4096);
  castN(w_uq,   wuq16,   512L * 2048);
  castN(w_o,    wo16,    2048L * 2048);
  castN(w_ff1,  wff1x,   2048L * 8192);
  castN(w_ff2,  wff2x,   8192L * 2048);

  // -------- LN1 --------
  layernorm_f16<<<(int)R, 256, 0, stream>>>(x, ln1_scale, ln1_bias, h16, 2048);

  // -------- projections from h --------
  gemm(h16, 2048, wdown16, b_down, nullptr, dkv16, nullptr, (int)R, 1024, 2048, 0);
  gemm64_f16_wmma<<<dim3(1, (int)R / 64), 256, 0, stream>>>(
      h16, 2048, wkr16, b_kr, krlin, (int)R, 64, 2048);
  gemm(h16, 2048, wqr16, b_qr, qrlin, nullptr, nullptr, (int)R, 1024, 2048, 0);

  // -------- RoPE (in place; buffers fully rewritten by GEMMs each call) ----
  rope_inplace<<<2048, 256, 0, stream>>>(krlin, 64, R);
  rope_inplace<<<2048, 256, 0, stream>>>(qrlin, 1024, R);
  castN(krlin, kr16, R * 64);
  castN(qrlin, qr16, R * 1024);

  // -------- latent up-projections (read dkv16 slices in place) --------
  gemm(dkv16 + 0,   1024, wukv16, b_ukv, nullptr, kv16, nullptr, (int)R, 4096, 512, 0);
  gemm(dkv16 + 512, 1024, wuq16,  b_uq,  nullptr, qb16, nullptr, (int)R, 2048, 512, 0);

  // -------- attention (flash, WMMA) --------
  mla_flash_attn<<<dim3(32, 16, 2), 256, 0, stream>>>(qb16, qr16, kv16, kr16,
                                                      attn16);

  // -------- output projection + residual --------
  gemm(attn16, 2048, wo16, b_o, x1, nullptr, x, (int)R, 2048, 2048, FLAG_RESID);

  // -------- LN2 + FFN + residual --------
  layernorm_f16<<<(int)R, 256, 0, stream>>>(x1, ln2_scale, ln2_bias, h2_16, 2048);
  gemm(h2_16, 2048, wff1x, b_ff1, nullptr, ff1_16, nullptr, (int)R, 8192, 2048, FLAG_GELU);
  gemm(ff1_16, 8192, wff2x, b_ff2, out, nullptr, x1, (int)R, 2048, 8192, FLAG_RESID);
}
